// GeoTransformerBaseline_56831007261102
// MI455X (gfx1250) — compile-verified, matrix operand order checked by reading, not run on hardware
//
#include <hip/hip_runtime.h>
#include <math.h>

// GeoTransformer forward for MI455X (gfx1250, wave32, WMMA bf16 16x16x32).
// All big GEMMs run on v_wmma_f32_16x16x32_bf16; rpe kept bf16 (L2-resident).

#define HIDD 256
#define FFND 512

typedef __attribute__((ext_vector_type(16))) __bf16 v16bf;
typedef __attribute__((ext_vector_type(8)))  float  v8f;

struct FragBF  { union { v16bf v; __bf16 e[16]; unsigned u[8]; }; };
struct FragAcc { union { v8f   v; float  f[8]; }; };

__device__ __forceinline__ unsigned short to_bf16(float x) {
  return __builtin_bit_cast(unsigned short, (__bf16)x);   // hw cvt, RNE
}

// ---- WMMA fragment loaders -------------------------------------------------
// 16-bit A/B 16x32 layout: lane l -> row/col = l&15, khalf = l>>4.
// VGPR0..3 hold K = khalf*8 + {0..7}; VGPR4..7 hold K = 16 + khalf*8 + {0..7}.
__device__ __forceinline__ void load_frag(FragBF& f, const float* __restrict__ p, int kh) {
  const float4* p0 = reinterpret_cast<const float4*>(p + kh * 8);
  const float4* p1 = reinterpret_cast<const float4*>(p + 16 + kh * 8);
  float4 a0 = p0[0], a1 = p0[1], b0 = p1[0], b1 = p1[1];
  f.e[0]  = (__bf16)a0.x; f.e[1]  = (__bf16)a0.y; f.e[2]  = (__bf16)a0.z; f.e[3]  = (__bf16)a0.w;
  f.e[4]  = (__bf16)a1.x; f.e[5]  = (__bf16)a1.y; f.e[6]  = (__bf16)a1.z; f.e[7]  = (__bf16)a1.w;
  f.e[8]  = (__bf16)b0.x; f.e[9]  = (__bf16)b0.y; f.e[10] = (__bf16)b0.z; f.e[11] = (__bf16)b0.w;
  f.e[12] = (__bf16)b1.x; f.e[13] = (__bf16)b1.y; f.e[14] = (__bf16)b1.z; f.e[15] = (__bf16)b1.w;
}
__device__ __forceinline__ void load_frag(FragBF& f, const unsigned short* __restrict__ p, int kh) {
  const uint4* p0 = reinterpret_cast<const uint4*>(p + kh * 8);
  const uint4* p1 = reinterpret_cast<const uint4*>(p + 16 + kh * 8);
  uint4 a = p0[0], b = p1[0];
  f.u[0] = a.x; f.u[1] = a.y; f.u[2] = a.z; f.u[3] = a.w;
  f.u[4] = b.x; f.u[5] = b.y; f.u[6] = b.z; f.u[7] = b.w;
}

__device__ __forceinline__ v8f wmma_bf16(const FragBF& a, const FragBF& b, v8f c) {
  return __builtin_amdgcn_wmma_f32_16x16x32_bf16(false, a.v, false, b.v, (short)0, c, false, false);
}

// ---- Generic batched WMMA GEMM --------------------------------------------
// C[batch] = A[batch] (M x K, row-major, lda) * B[batch] (K-contiguous columns,
// column col at B + col*ldb). One 16x16 tile per wave, 8 waves per block.
template <typename TA, typename TB>
__global__ __launch_bounds__(256)
void gemm_wmma_kernel(const TA* __restrict__ Abase, int lda, long aS,
                      const TB* __restrict__ Bbase, int ldb, long bS,
                      float* __restrict__ Cbase, int ldc, long cS,
                      const float* __restrict__ bias,
                      int M, int Ncols, int NValid, int K,
                      int relu, int transOut) {
  const int wave = threadIdx.x >> 5;
  const int lane = threadIdx.x & 31;
  const int nTilesN = Ncols >> 4;
  const int total = (M >> 4) * nTilesN;
  int tile = blockIdx.x * (blockDim.x >> 5) + wave;
  if (tile >= total) return;                 // wave-uniform: EXEC stays full
  const int tN = tile % nTilesN;
  const int tM = tile / nTilesN;
  const long batch = blockIdx.y;
  const TA* A = Abase + batch * aS;
  const TB* B = Bbase + batch * bS;
  float*    C = Cbase + batch * cS;
  const int kh = lane >> 4;
  const int l15 = lane & 15;
  const int rowg = tM * 16 + l15;
  const int colg = tN * 16 + l15;
  const int bcol = (colg < NValid) ? colg : 0;
  const TA* aRow = A + (long)rowg * lda;
  const TB* bCol = B + (long)bcol * ldb;
  FragAcc acc;
#pragma unroll
  for (int i = 0; i < 8; ++i) acc.f[i] = 0.0f;
  for (int kb = 0; kb < K; kb += 32) {
    FragBF fa, fb;
    load_frag(fa, aRow + kb, kh);
    load_frag(fb, bCol + kb, kh);
    acc.v = wmma_bf16(fa, fb, acc.v);
  }
  const float bv = bias ? bias[bcol] : 0.0f;
#pragma unroll
  for (int i = 0; i < 8; ++i) {
    float r = acc.f[i] + bv;
    if (relu) r = fmaxf(r, 0.0f);
    const int rg = tM * 16 + kh * 8 + i;     // C: VGPR i -> row i (+8 for hi half)
    if (transOut) C[(long)colg * ldc + rg] = r;
    else          C[(long)rg   * ldc + colg] = r;
  }
}

template <typename TA, typename TB>
static void gemm(hipStream_t s, const TA* A, int lda, long aS,
                 const TB* B, int ldb, long bS,
                 float* C, int ldc, long cS, const float* bias,
                 int M, int Ncols, int NValid, int K, int relu, int transOut, int batch) {
  int tiles = (M >> 4) * (Ncols >> 4);
  dim3 grid((tiles + 7) / 8, batch, 1);
  gemm_wmma_kernel<TA, TB><<<grid, 256, 0, s>>>(A, lda, aS, B, ldb, bS, C, ldc, cS,
                                                bias, M, Ncols, NValid, K, relu, transOut);
}

// ---- Geometry: pairwise dist indices, 3-NN, angle indices ------------------
__global__ __launch_bounds__(256)
void knn_geom_kernel(const float* __restrict__ pts,
                     float* __restrict__ d_ind, float* __restrict__ a_ind) {
  __shared__ float sDist[256];
  __shared__ float rv[256];
  __shared__ int   ri[256];
  __shared__ int   nn[3];
  const int n = blockIdx.x, m = threadIdx.x;
  const float pnx = pts[n * 3 + 0], pny = pts[n * 3 + 1], pnz = pts[n * 3 + 2];
  const float pmx = pts[m * 3 + 0], pmy = pts[m * 3 + 1], pmz = pts[m * 3 + 2];
  const float dx = pnx - pmx, dy = pny - pmy, dz = pnz - pmz;  // diffs[n,m] = p[n]-p[m]
  const float dist = sqrtf(dx * dx + dy * dy + dz * dz);
  d_ind[n * 256 + m] = dist * 5.0f;                            // / SIGMA_D(0.2)
  sDist[m] = (m == n) ? 3.0e38f : dist;
  __syncthreads();
  for (int k = 0; k < 3; ++k) {
    float v = sDist[m];
    for (int j = 0; j < k; ++j) if (nn[j] == m) v = 3.0e38f;
    rv[m] = v; ri[m] = m;
    __syncthreads();
    for (int s = 128; s > 0; s >>= 1) {
      if (m < s) { if (rv[m + s] < rv[m]) { rv[m] = rv[m + s]; ri[m] = ri[m + s]; } }
      __syncthreads();
    }
    if (m == 0) nn[k] = ri[0];
    __syncthreads();
  }
#pragma unroll
  for (int k = 0; k < 3; ++k) {
    const int idx = nn[k];
    const float rx = pts[idx * 3 + 0] - pnx;
    const float ry = pts[idx * 3 + 1] - pny;
    const float rz = pts[idx * 3 + 2] - pnz;
    const float cx = ry * dz - rz * dy;
    const float cy = rz * dx - rx * dz;
    const float cz = rx * dy - ry * dx;
    const float sv = sqrtf(cx * cx + cy * cy + cz * cz);
    const float cv = rx * dx + ry * dy + rz * dz;
    a_ind[(n * 256 + m) * 3 + k] = atan2f(sv, cv) * 3.81971863420549f; // 180/(15*pi)
  }
}

// ---- PE generation fused with projection GEMM + K=3 max --------------------
// isCos is a compile-time constant per unrolled K-step (the 128 boundary is
// 32-aligned), so no branches; fr[] is shared across the 4 matrix fragments.
__device__ __forceinline__ void pe_frag(FragBF& f, float x, const float* fr, bool isCos) {
#pragma unroll
  for (int i = 0; i < 16; ++i) {
    const float a = x * fr[i];
    f.e[i] = (__bf16)(isCos ? __cosf(a) : __sinf(a));
  }
}

__global__ __launch_bounds__(256)
void pe_project_kernel(const float* __restrict__ d_ind,
                       const float* __restrict__ a_ind,
                       const unsigned short* __restrict__ WdT,  // [c][k] bf16
                       const unsigned short* __restrict__ WaT,  // [c][k] bf16
                       unsigned short* __restrict__ rpe) {      // [(n*256+m)][c] bf16
  const int wave = threadIdx.x >> 5, lane = threadIdx.x & 31;
  int tile = blockIdx.x * (blockDim.x >> 5) + wave;
  if (tile >= 65536) return;                 // 4096 row tiles x 16 col tiles
  const int tN = tile & 15, tM = tile >> 4;
  const int kh = lane >> 4, l15 = lane & 15;
  const int rg = tM * 16 + l15;
  const int colg = tN * 16 + l15;
  const float xd  = d_ind[rg];
  const float xa0 = a_ind[rg * 3 + 0];
  const float xa1 = a_ind[rg * 3 + 1];
  const float xa2 = a_ind[rg * 3 + 2];
  const unsigned short* bd = WdT + (long)colg * 256;
  const unsigned short* ba = WaT + (long)colg * 256;
  FragAcc cd, c0, c1, c2;
#pragma unroll
  for (int i = 0; i < 8; ++i) { cd.f[i] = 0.f; c0.f[i] = 0.f; c1.f[i] = 0.f; c2.f[i] = 0.f; }
  const float L2 = 0.1038102532f;            // log2(10000)/128
#pragma unroll
  for (int kb = 0; kb < 256; kb += 32) {
    const bool isCos = (kb >= 128);          // uniform per K-step after unroll
    float fr[16];
#pragma unroll
    for (int j = 0; j < 8; ++j) {
      fr[j]     = __builtin_amdgcn_exp2f(-(float)((kb + kh * 8 + j) & 127) * L2);
      fr[8 + j] = __builtin_amdgcn_exp2f(-(float)((kb + 16 + kh * 8 + j) & 127) * L2);
    }
    FragBF fb, fa;
    load_frag(fb, bd + kb, kh);
    pe_frag(fa, xd, fr, isCos);  cd.v = wmma_bf16(fa, fb, cd.v);
    load_frag(fb, ba + kb, kh);
    pe_frag(fa, xa0, fr, isCos); c0.v = wmma_bf16(fa, fb, c0.v);
    pe_frag(fa, xa1, fr, isCos); c1.v = wmma_bf16(fa, fb, c1.v);
    pe_frag(fa, xa2, fr, isCos); c2.v = wmma_bf16(fa, fb, c2.v);
  }
#pragma unroll
  for (int i = 0; i < 8; ++i) {
    float r = cd.f[i] + fmaxf(fmaxf(c0.f[i], c1.f[i]), c2.f[i]);
    const int rr = tM * 16 + kh * 8 + i;
    rpe[(long)rr * 256 + colg] = to_bf16(r);
  }
}

// ---- Softmax over m (one wave per (h,n) row) -------------------------------
__global__ __launch_bounds__(256)
void softmax_kernel(const float* __restrict__ S, const float* __restrict__ Ppad,
                    float* __restrict__ out, int nRows, float scale) {
  const int wave = threadIdx.x >> 5, lane = threadIdx.x & 31;
  const int r = blockIdx.x * (blockDim.x >> 5) + wave;
  if (r >= nRows) return;
  const int h = r >> 8, n = r & 255;
  const float* row = S + (long)r * 256;
  float v[8]; float mx = -3.0e38f;
#pragma unroll
  for (int j = 0; j < 8; ++j) {
    const int m = lane + j * 32;
    float t = row[m];
    if (Ppad) t += Ppad[((long)(n * 256 + m)) * 16 + h];
    t *= scale;
    v[j] = t; mx = fmaxf(mx, t);
  }
  for (int s = 16; s > 0; s >>= 1) mx = fmaxf(mx, __shfl_xor(mx, s, 32));
  float sum = 0.f;
#pragma unroll
  for (int j = 0; j < 8; ++j) { v[j] = __expf(v[j] - mx); sum += v[j]; }
  for (int s = 16; s > 0; s >>= 1) sum += __shfl_xor(sum, s, 32);
  const float inv = 1.0f / sum;
#pragma unroll
  for (int j = 0; j < 8; ++j) out[(long)r * 256 + lane + j * 32] = v[j] * inv;
}

// ---- out = LayerNorm(a + b) * g + beta (one wave per row) ------------------
__global__ __launch_bounds__(256)
void add_ln_kernel(const float* __restrict__ a, const float* __restrict__ b,
                   const float* __restrict__ g, const float* __restrict__ beta,
                   float* __restrict__ out, int nRows) {
  const int wave = threadIdx.x >> 5, lane = threadIdx.x & 31;
  const int r = blockIdx.x * (blockDim.x >> 5) + wave;
  if (r >= nRows) return;
  float v[8]; float s = 0.f;
#pragma unroll
  for (int j = 0; j < 8; ++j) {
    const int c = lane + j * 32;
    const float t = a[(long)r * 256 + c] + b[(long)r * 256 + c];
    v[j] = t; s += t;
  }
  for (int m = 16; m > 0; m >>= 1) s += __shfl_xor(s, m, 32);
  const float mean = s * (1.0f / 256.0f);
  float s2 = 0.f;
#pragma unroll
  for (int j = 0; j < 8; ++j) { const float d = v[j] - mean; s2 += d * d; }
  for (int m = 16; m > 0; m >>= 1) s2 += __shfl_xor(s2, m, 32);
  const float rstd = rsqrtf(s2 * (1.0f / 256.0f) + 1e-5f);
#pragma unroll
  for (int j = 0; j < 8; ++j) {
    const int c = lane + j * 32;
    out[(long)r * 256 + c] = (v[j] - mean) * rstd * g[c] + beta[c];
  }
}

// ---- Weight prep -----------------------------------------------------------
__global__ void transpose_cast_kernel(const float* __restrict__ src,
                                      unsigned short* __restrict__ dst,
                                      int rows, int cols) {
  const int i = blockIdx.x * blockDim.x + threadIdx.x;
  if (i >= rows * cols) return;
  const int r = i / cols, c = i % cols;
  dst[(long)c * rows + r] = to_bf16(src[i]);
}
__global__ void cast_kernel(const float* __restrict__ src,
                            unsigned short* __restrict__ dst, int n) {
  const int i = blockIdx.x * blockDim.x + threadIdx.x;
  if (i < n) dst[i] = to_bf16(src[i]);
}

// ============================================================================
extern "C" void kernel_launch(void* const* d_in, const int* in_sizes, int n_in,
                              void* d_out, int out_size, void* d_ws, size_t ws_size,
                              hipStream_t stream) {
  (void)in_sizes; (void)n_in; (void)out_size; (void)ws_size;
  const float* ref_pts   = (const float*)d_in[0];
  const float* src_pts   = (const float*)d_in[1];
  const float* ref_feats = (const float*)d_in[2];
  const float* src_feats = (const float*)d_in[3];

  // ---- parameter walk: jax pytree flattening (dict keys sorted) ----
  int pi = 4;
  auto nextp = [&]() { return (const float*)d_in[pi++]; };
  struct Lin { const float* b; const float* w; };
  struct Layer {
    Lin k, o, q, v; const float* rw;
    Lin f1, f2;
    const float *n1b, *n1g, *n2b, *n2g;
    unsigned short *kT, *oT, *qT, *vT, *rC, *f1T, *f2T;
  };
  Layer selfL[3], crossL[3];
  for (int b = 0; b < 3; ++b) {
    Layer& c = crossL[b];                      // 'cross' < 'self'
    c.k.b = nextp(); c.k.w = nextp(); c.o.b = nextp(); c.o.w = nextp();
    c.q.b = nextp(); c.q.w = nextp(); c.v.b = nextp(); c.v.w = nextp();
    c.rw = nullptr;
    c.f1.b = nextp(); c.f1.w = nextp(); c.f2.b = nextp(); c.f2.w = nextp();
    c.n1b = nextp(); c.n1g = nextp(); c.n2b = nextp(); c.n2g = nextp();
    Layer& s = selfL[b];                       // attn keys: k,o,q,r,v
    s.k.b = nextp(); s.k.w = nextp(); s.o.b = nextp(); s.o.w = nextp();
    s.q.b = nextp(); s.q.w = nextp();
    s.rw  = nextp();
    s.v.b = nextp(); s.v.w = nextp();
    s.f1.b = nextp(); s.f1.w = nextp(); s.f2.b = nextp(); s.f2.w = nextp();
    s.n1b = nextp(); s.n1g = nextp(); s.n2b = nextp(); s.n2g = nextp();
  }
  const float* proj_a_w = nextp();             // 'proj_a' < 'proj_d'
  const float* proj_d_w = nextp();

  // ---- workspace layout ----
  char* wsBase = (char*)d_ws;
  size_t off = 0;
  auto alloc  = [&](size_t bytes) { void* r = wsBase + off; off = (off + bytes + 255) & ~(size_t)255; return r; };
  auto allocF = [&](size_t n) { return (float*)alloc(n * 4); };
  auto allocH = [&](size_t n) { return (unsigned short*)alloc(n * 2); };

  unsigned short* projdT = allocH(65536);
  unsigned short* projaT = allocH(65536);
  float* dIndR = allocF(65536);   float* dIndS = allocF(65536);
  float* aIndR = allocF(196608);  float* aIndS = allocF(196608);
  unsigned short* rpeR = allocH(16777216);
  unsigned short* rpeS = allocH(16777216);
  float* Qb = allocF(65536);
  float* Kb = allocF(65536);
  float* Vt = allocF(65536);      // V stored transposed: [c][n]
  float* Ob = allocF(65536);
  float* T1 = allocF(65536);
  float* X1 = allocF(65536);
  float* Hf = allocF(131072);
  float* Qr = allocF(262144);     // [n][h][c]
  float* Sc = allocF(262144);     // attn_c [h][n][m]
  float* Sp = allocF(1048576);    // attn_p padded [n][m][16]
  float* Pb = allocF(262144);     // softmax probs [h][n][m]
  float* xR  = allocF(65536);  float* xS  = allocF(65536);
  float* xR2 = allocF(65536);  float* xS2 = allocF(65536);

  // ---- weight prep (transpose+cast to bf16; rw plain cast) ----
  auto tc = [&](const float* w, unsigned short* dst, int din, int dout) {
    const int n = din * dout;
    transpose_cast_kernel<<<(n + 255) / 256, 256, 0, stream>>>(w, dst, din, dout);
  };
  for (int b = 0; b < 3; ++b) {
    Layer* two[2] = { &selfL[b], &crossL[b] };
    for (int t = 0; t < 2; ++t) {
      Layer& L = *two[t];
      L.kT  = allocH(65536);  tc(L.k.w,  L.kT,  256, 256);
      L.oT  = allocH(65536);  tc(L.o.w,  L.oT,  256, 256);
      L.qT  = allocH(65536);  tc(L.q.w,  L.qT,  256, 256);
      L.vT  = allocH(65536);  tc(L.v.w,  L.vT,  256, 256);
      L.f1T = allocH(131072); tc(L.f1.w, L.f1T, 256, 512);
      L.f2T = allocH(131072); tc(L.f2.w, L.f2T, 512, 256);
      if (L.rw) { L.rC = allocH(65536); cast_kernel<<<256, 256, 0, stream>>>(L.rw, L.rC, 65536); }
      else L.rC = nullptr;
    }
  }
  tc(proj_d_w, projdT, 256, 256);
  tc(proj_a_w, projaT, 256, 256);

  // ---- geometry + geometric embedding (rpe, bf16) ----
  knn_geom_kernel<<<256, 256, 0, stream>>>(ref_pts, dIndR, aIndR);
  knn_geom_kernel<<<256, 256, 0, stream>>>(src_pts, dIndS, aIndS);
  pe_project_kernel<<<8192, 256, 0, stream>>>(dIndR, aIndR, projdT, projaT, rpeR);
  pe_project_kernel<<<8192, 256, 0, stream>>>(dIndS, aIndS, projdT, projaT, rpeS);

  // ---- one attention layer (self if rpe != null, else cross) ----
  auto run_layer = [&](const float* xq, const float* xkv, const unsigned short* rpe,
                       Layer& L, float* xOut) {
    const bool hasR = (L.rC != nullptr) && (rpe != nullptr);
    gemm<float, unsigned short>(stream, xq,  HIDD, 0, L.qT, HIDD, 0, Qb, HIDD, 0, L.q.b, 256, 256, 256, 256, 0, 0, 1);
    gemm<float, unsigned short>(stream, xkv, HIDD, 0, L.kT, HIDD, 0, Kb, HIDD, 0, L.k.b, 256, 256, 256, 256, 0, 0, 1);
    gemm<float, unsigned short>(stream, xkv, HIDD, 0, L.vT, HIDD, 0, Vt, HIDD, 0, L.v.b, 256, 256, 256, 256, 0, 1, 1);
    // attn_c[h,n,m] = Q_h . K_h   (batch over heads: A/B offset by h*64)
    gemm<float, float>(stream, Qb, HIDD, 64, Kb, HIDD, 64, Sc, 256, 65536, nullptr, 256, 256, 256, 64, 0, 0, 4);
    if (hasR) {
      // Qr[n,h,c] = sum_d Q[n,h,d] * Wr[c, h*64+d]
      gemm<float, unsigned short>(stream, Qb, HIDD, 64, L.rC, HIDD, 64, Qr, 1024, 256, nullptr, 256, 256, 256, 64, 0, 0, 4);
      // attn_p[n,m,h] = sum_c rpe[n,m,c] * Qr[n,h,c]   (batch over n, Ncols padded 16)
      gemm<unsigned short, float>(stream, rpe, HIDD, 65536, Qr, HIDD, 1024, Sp, 16, 4096,
                                  nullptr, 256, 16, 4, 256, 0, 0, 256);
    }
    softmax_kernel<<<128, 256, 0, stream>>>(Sc, hasR ? Sp : nullptr, Pb, 1024, 0.125f);
    // O[n, h*64+d] = sum_m P[h,n,m] * Vt[h*64+d, m]
    gemm<float, float>(stream, Pb, 256, 65536, Vt, 256, 16384, Ob, HIDD, 64, nullptr, 256, 64, 64, 256, 0, 0, 4);
    gemm<float, unsigned short>(stream, Ob, HIDD, 0, L.oT, HIDD, 0, T1, HIDD, 0, L.o.b, 256, 256, 256, 256, 0, 0, 1);
    add_ln_kernel<<<32, 256, 0, stream>>>(T1, xq, L.n1g, L.n1b, X1, 256);
    gemm<float, unsigned short>(stream, X1, HIDD, 0, L.f1T, HIDD, 0, Hf, FFND, 0, L.f1.b, 256, 512, 512, 256, 1, 0, 1);
    gemm<float, unsigned short>(stream, Hf, FFND, 0, L.f2T, FFND, 0, T1, HIDD, 0, L.f2.b, 256, 256, 256, 512, 0, 0, 1);
    add_ln_kernel<<<32, 256, 0, stream>>>(T1, X1, L.n2g, L.n2b, xOut, 256);
  };

  float* outRef = (float*)d_out;
  float* outSrc = (float*)d_out + 65536;
  const float* curR = ref_feats;
  const float* curS = src_feats;
  for (int b = 0; b < 3; ++b) {
    run_layer(curR, curR, rpeR, selfL[b], xR); curR = xR;
    run_layer(curS, curS, rpeS, selfL[b], xS); curS = xS;
    float* oR = (b == 2) ? outRef : xR2;
    float* oS = (b == 2) ? outSrc : xS2;
    run_layer(curR, curS, nullptr, crossL[b], oR);
    run_layer(curS, curR, nullptr, crossL[b], oS);
    curR = oR; curS = oS;
  }
}
